// MoELayer_16947940950328
// MI455X (gfx1250) — matrix operand order repaired and checked by reading, not compiled, not run on hardware
//
#include <hip/hip_runtime.h>

#ifndef __has_builtin
#define __has_builtin(x) 0
#endif

typedef __attribute__((ext_vector_type(16))) _Float16 v16h;
typedef __attribute__((ext_vector_type(8)))  _Float16 v8h;
typedef __attribute__((ext_vector_type(4)))  _Float16 v4h;
typedef __attribute__((ext_vector_type(8)))  float    v8f;
typedef __attribute__((ext_vector_type(4)))  float    v4f;
typedef __attribute__((ext_vector_type(4)))  int      v4i;
typedef __attribute__((ext_vector_type(8)))  int      v8i;
typedef __attribute__((ext_vector_type(4)))  unsigned v4u;

constexpr int T  = 4096;   // B*L tokens
constexpr int Dq = 1024;
constexpr int Fq = 1024;
constexpr int E  = 8;
constexpr int LDK = 72;    // padded LDS K stride (f16): 64 + 8 pad = 144B rows

// ---- gfx1250 async global->LDS path (probe-confirmed on this toolchain) ---
#if __has_builtin(__builtin_amdgcn_global_load_async_to_lds_b128)
#define HAVE_ASYNC_LDS 1
#endif
// ---- gfx1250 Tensor Data Mover (6-arg form on this toolchain) -------------
#if __has_builtin(__builtin_amdgcn_tensor_load_to_lds)
#define HAVE_TDM 1
#endif

__device__ __forceinline__ void copy_b128_to_lds(const _Float16* g, _Float16* l) {
#ifdef HAVE_ASYNC_LDS
    __builtin_amdgcn_global_load_async_to_lds_b128(
        (__attribute__((address_space(1))) v4i*)g,
        (__attribute__((address_space(3))) v4i*)l, 0, 0);
#else
    *(v8h*)l = *(const v8h*)g;
#endif
}

__device__ __forceinline__ void wait_async_copies() {
#ifdef HAVE_ASYNC_LDS
#if __has_builtin(__builtin_amdgcn_s_wait_asynccnt)
    __builtin_amdgcn_s_wait_asynccnt(0);
#else
    asm volatile("s_wait_asynccnt 0x0" ::: "memory");
#endif
#endif
}

__device__ __forceinline__ void wait_tensor_copies() {
#if __has_builtin(__builtin_amdgcn_s_wait_tensorcnt)
    __builtin_amdgcn_s_wait_tensorcnt(0);
#else
    asm volatile("s_wait_tensorcnt 0x0" ::: "memory");
#endif
}

#ifdef HAVE_TDM
// 2D TDM tile load: tile_dim0=64 f16 (128B rows), tile_dim1=128 rows,
// LDS rows padded by 16B (pad_interval=32 DWORDs, pad_amount=4 DWORDs -> LDK=72)
// D# packing per CDNA5 ISA ch.8 (group0: ctrl/lds/global/type, group1: dims).
__device__ __forceinline__ void tdm_load_tile_2d(const _Float16* gbase,
                                                 unsigned lds_off,
                                                 int row_stride_elems,
                                                 int tensor_rows) {
    unsigned long long ga = (unsigned long long)(size_t)gbase;
    v4u g0;
    g0[0] = 1u;                                          // count=1, user D#
    g0[1] = lds_off;                                     // lds_addr (bytes)
    g0[2] = (unsigned)ga;                                // global_addr[31:0]
    g0[3] = (unsigned)((ga >> 32) & 0x01FFFFFFu) | (2u << 30);  // addr[56:32] | type=2
    v8i g1;
    g1[0] = (int)((1u << 16)      // data_size = 1 (2 bytes)
                | (1u << 20)      // pad_enable
                | (4u << 22)      // pad_interval: 32 DWORDs
                | (3u << 25));    // pad_amount: 4 DWORDs (16B)
    unsigned td0 = (unsigned)row_stride_elems;           // tensor_dim0
    unsigned td1 = (unsigned)tensor_rows;                // tensor_dim1
    g1[1] = (int)((td0 & 0xFFFFu) << 16);
    g1[2] = (int)(((td0 >> 16) & 0xFFFFu) | ((td1 & 0xFFFFu) << 16));
    g1[3] = (int)(((td1 >> 16) & 0xFFFFu) | (64u << 16));  // | tile_dim0 = 64
    g1[4] = 128;                                         // tile_dim1 = 128 rows
    g1[5] = row_stride_elems;                            // tensor_dim0_stride lo32
    g1[6] = 0;
    g1[7] = 0;
    v4i z4 = {0, 0, 0, 0};
    v8i z8 = {0, 0, 0, 0, 0, 0, 0, 0};
    __builtin_amdgcn_tensor_load_to_lds(g0, g1, z4, z4, z8, 0);
}
#endif

// ---------------------------------------------------------------------------
__device__ __forceinline__ v16h ld_v16h(const _Float16* p) {
    v8h lo = *(const v8h*)p;
    v8h hi = *(const v8h*)(p + 8);
    v16h r;
#pragma unroll
    for (int i = 0; i < 8; ++i) { r[i] = lo[i]; r[i + 8] = hi[i]; }
    return r;
}

__device__ __forceinline__ float fast_sigmoid(float x) {
#if __has_builtin(__builtin_amdgcn_rcpf)
    return __builtin_amdgcn_rcpf(1.f + __expf(-x));   // v_rcp_f32, no div fixup
#else
    return 1.f / (1.f + __expf(-x));
#endif
}

// ---------------------------------------------------------------------------
// 1) cast x (f32) -> x16 (f16)
// ---------------------------------------------------------------------------
__global__ void cast_x_kernel(const float* __restrict__ x,
                              _Float16* __restrict__ x16, int n4) {
    int i = blockIdx.x * blockDim.x + threadIdx.x;
    if (i >= n4) return;
    v4f v = ((const v4f*)x)[i];
    v4h h;
#pragma unroll
    for (int j = 0; j < 4; ++j) h[j] = (_Float16)v[j];
    ((v4h*)x16)[i] = h;
}

// ---------------------------------------------------------------------------
// 2) router: one wave per token; sigmoid gates + top-2 + normalize.
// ---------------------------------------------------------------------------
__global__ void router_kernel(const float* __restrict__ x,
                              const float* __restrict__ Wr,     // [D,E]
                              const float* __restrict__ bias,   // [E]
                              float* __restrict__ wgt,          // [T,E]
                              float* __restrict__ out_aux) {
    const int wave = threadIdx.x >> 5;
    const int lane = threadIdx.x & 31;
    const int t = blockIdx.x * 8 + wave;

    float acc[E];
#pragma unroll
    for (int e = 0; e < E; ++e) acc[e] = 0.f;

    const float* xr = x + (size_t)t * Dq;
    for (int d = lane; d < Dq; d += 32) {
        float xv = xr[d];
        v4f w0 = ((const v4f*)(Wr + (size_t)d * E))[0];
        v4f w1 = ((const v4f*)(Wr + (size_t)d * E))[1];
#pragma unroll
        for (int j = 0; j < 4; ++j) {
            acc[j]     += xv * w0[j];
            acc[4 + j] += xv * w1[j];
        }
    }
#pragma unroll
    for (int e = 0; e < E; ++e) {
#pragma unroll
        for (int m = 16; m >= 1; m >>= 1)
            acc[e] += __shfl_xor(acc[e], m, 32);
    }

    if (lane == 0) {
        float g[E];
#pragma unroll
        for (int e = 0; e < E; ++e)
            g[e] = fast_sigmoid(acc[e] + bias[e]);
        int i1 = 0; float g1 = g[0];
#pragma unroll
        for (int e = 1; e < E; ++e) if (g[e] > g1) { g1 = g[e]; i1 = e; }
        int i2 = -1; float g2 = -1.f;
#pragma unroll
        for (int e = 0; e < E; ++e)
            if (e != i1 && g[e] > g2) { g2 = g[e]; i2 = e; }
        float inv = 1.f / (g1 + g2 + 1e-9f);
#pragma unroll
        for (int e = 0; e < E; ++e) {
            float v = 0.f;
            if (e == i1) v = g1 * inv;
            if (e == i2) v = g2 * inv;
            wgt[(size_t)t * E + e] = v;
        }
    }
    if (blockIdx.x == 0 && threadIdx.x == 0) out_aux[0] = 0.f;  // aux_loss
}

// ---------------------------------------------------------------------------
// 3) deterministic per-expert token compaction (ordered block scan, no atomics)
// ---------------------------------------------------------------------------
__global__ void build_lists_kernel(const float* __restrict__ wgt,  // [T,E]
                                   int* __restrict__ cnt,          // [E]
                                   int* __restrict__ idx) {        // [E,T]
    const int e = blockIdx.x;
    const int tid = threadIdx.x;
    const int lane = tid & 31, wid = tid >> 5;
    __shared__ int sWs[8];
    __shared__ int sBase;
    if (tid == 0) sBase = 0;
    __syncthreads();
    for (int t0 = 0; t0 < T; t0 += 256) {
        int t = t0 + tid;
        bool sel = wgt[(size_t)t * E + e] > 0.f;
        unsigned mask = (unsigned)__ballot(sel);
        int lpre = __popc(mask & ((1u << lane) - 1u));
        if (lane == 0) sWs[wid] = __popc(mask);
        __syncthreads();
        int woff = 0, tot = 0;
#pragma unroll
        for (int i = 0; i < 8; ++i) {
            int c = sWs[i];
            if (i < wid) woff += c;
            tot += c;
        }
        int base = sBase;
        if (sel) idx[(size_t)e * T + base + woff + lpre] = t;
        __syncthreads();
        if (tid == 0) sBase = base + tot;
        __syncthreads();
    }
    if (tid == 0) cnt[e] = sBase;
}

// ---------------------------------------------------------------------------
// 4) fused gate+up GEMM on gathered token rows:
//    act[r,:] = silu(X[tok(r)]@Wg) * (X[tok(r)]@Wu)
//    block = 256 thr (8 waves, 4x2), tile 128(M)x64(N), K-stage 64.
// ---------------------------------------------------------------------------
__global__ __launch_bounds__(256)
void gemm_gateup_kernel(const _Float16* __restrict__ X16,  // [T,Dq] f16
                        const float* __restrict__ Wg,      // [Dq,Fq] f32
                        const float* __restrict__ Wu,      // [Dq,Fq] f32
                        _Float16* __restrict__ act,        // [T,Fq] f16 (compacted)
                        const int* __restrict__ idx,       // token list or null
                        const int* __restrict__ cntp) {    // count or null (=T)
    const int cnt = cntp ? cntp[0] : T;
    const int m0 = blockIdx.y * 128;
    if (m0 >= cnt) return;
    const int n0 = blockIdx.x * 64;

    __shared__ __align__(16) _Float16 sA [128][LDK];
    __shared__ __align__(16) _Float16 sBg[64][LDK];   // [n][k]
    __shared__ __align__(16) _Float16 sBu[64][LDK];
    __shared__ int sTok[128];

    const int tid  = threadIdx.x;
    const int lane = tid & 31;
    const int wid  = tid >> 5;
    const int wm   = (wid & 3) * 32;
    const int wn   = (wid >> 2) * 32;

    if (tid < 128) {
        int r  = m0 + tid;
        int rc = r < cnt ? r : cnt - 1;     // clamp padding rows
        sTok[tid] = idx ? idx[rc] : rc;
    }

    v8f cg[2][2] = {}, cu[2][2] = {};

    for (int k0 = 0; k0 < Dq; k0 += 64) {
        __syncthreads();
        if (k0 + 64 < Dq) {   // prefetch next weight K-rows (global_prefetch_b8)
            __builtin_prefetch(Wg + (size_t)(k0 + 64 + (tid >> 2)) * Fq + n0, 0, 0);
            __builtin_prefetch(Wu + (size_t)(k0 + 64 + (tid >> 2)) * Fq + n0, 0, 0);
        }
        // A tile: 128 x 64 f16 gathered rows (async global->LDS)
#pragma unroll
        for (int i = 0; i < 4; ++i) {
            int c   = tid + 256 * i;
            int row = c >> 3;
            int ko  = (c & 7) * 8;
            copy_b128_to_lds(X16 + (size_t)sTok[row] * Dq + k0 + ko, &sA[row][ko]);
        }
        // B tiles: 64(K) x 64(N) f32, cvt f16, store transposed [n][k]
#pragma unroll
        for (int i = 0; i < 4; ++i) {
            int c  = tid + 256 * i;
            int kk = c >> 4;
            int nc = (c & 15) * 4;
            size_t off = (size_t)(k0 + kk) * Fq + n0 + nc;
            v4f g4 = *(const v4f*)(Wg + off);
            v4f u4 = *(const v4f*)(Wu + off);
#pragma unroll
            for (int j = 0; j < 4; ++j) {
                sBg[nc + j][kk] = (_Float16)g4[j];
                sBu[nc + j][kk] = (_Float16)u4[j];
            }
        }
        wait_async_copies();
        __syncthreads();

#pragma unroll
        for (int ks = 0; ks < 64; ks += 32) {
            v16h af[2], bg[2], bu[2];
#pragma unroll
            for (int tm = 0; tm < 2; ++tm) {
                int row = wm + tm * 16 + (lane & 15);
                int kh  = ks + (lane >> 4) * 16;
                v8h lo = *(const v8h*)&sA[row][kh];
                v8h hi = *(const v8h*)&sA[row][kh + 8];
                v16h a;
#pragma unroll
                for (int j = 0; j < 8; ++j) { a[j] = lo[j]; a[j + 8] = hi[j]; }
                af[tm] = a;
            }
#pragma unroll
            for (int tn = 0; tn < 2; ++tn) {
                int nn = wn + tn * 16 + (lane & 15);
                int kh = ks + (lane >> 4) * 16;
                bg[tn] = ld_v16h(&sBg[nn][kh]);
                bu[tn] = ld_v16h(&sBu[nn][kh]);
            }
#pragma unroll
            for (int tm = 0; tm < 2; ++tm)
#pragma unroll
                for (int tn = 0; tn < 2; ++tn) {
                    cg[tm][tn] = __builtin_amdgcn_wmma_f32_16x16x32_f16(
                        false, af[tm], false, bg[tn], (short)0, cg[tm][tn], false, false);
                    cu[tm][tn] = __builtin_amdgcn_wmma_f32_16x16x32_f16(
                        false, af[tm], false, bu[tn], (short)0, cu[tm][tn], false, false);
                }
        }
    }

#pragma unroll
    for (int tm = 0; tm < 2; ++tm)
#pragma unroll
        for (int tn = 0; tn < 2; ++tn)
#pragma unroll
            for (int r = 0; r < 8; ++r) {
                int row = m0 + wm + tm * 16 + (lane >> 4) * 8 + r;  // compacted row
                int col = n0 + wn + tn * 16 + (lane & 15);
                float hg = cg[tm][tn][r];
                float hu = cu[tm][tn][r];
                act[(size_t)row * Fq + col] = (_Float16)(hg * fast_sigmoid(hg) * hu);
            }
}

// ---------------------------------------------------------------------------
// 5) down GEMM + scatter-combine: out[tok(r)] (+)= gate * (act[r] @ Wd)
//    A tile is a dense 2D tile -> loaded by the Tensor Data Mover (one
//    tensor_load_to_lds per K-stage, LDS padding via D# pad fields).
// ---------------------------------------------------------------------------
__global__ __launch_bounds__(256)
void gemm_down_kernel(const _Float16* __restrict__ A16,  // [*,Fq] f16 compacted
                      const float* __restrict__ Wd,      // [Fq,Dq] f32
                      const float* __restrict__ wgt,     // [T,E]
                      int e,                             // -1 => shared expert
                      const int* __restrict__ idx,
                      const int* __restrict__ cntp,
                      float* __restrict__ out,           // [T,Dq]
                      int init) {
    const int cnt = cntp ? cntp[0] : T;
    const int m0 = blockIdx.y * 128;
    if (m0 >= cnt) return;
    const int n0 = blockIdx.x * 64;

    __shared__ __align__(16) _Float16 sA[128][LDK];
    __shared__ __align__(16) _Float16 sB[64][LDK];
    __shared__ int   sTok[128];
    __shared__ float sScl[128];

    const int tid  = threadIdx.x;
    const int lane = tid & 31;
    const int wid  = tid >> 5;
    const int wm   = (wid & 3) * 32;
    const int wn   = (wid >> 2) * 32;

    if (tid < 128) {
        int r = m0 + tid;
        if (r < cnt) {
            int tok = idx ? idx[r] : r;
            sTok[tid] = tok;
            sScl[tid] = (e >= 0) ? wgt[(size_t)tok * E + e] : 1.f;
        } else {
            sTok[tid] = -1;
            sScl[tid] = 0.f;
        }
    }

#ifdef HAVE_TDM
    const unsigned sA_off =
        (unsigned)(size_t)(__attribute__((address_space(3))) void*)&sA[0][0];
#endif

    v8f cc[2][2] = {};

    for (int k0 = 0; k0 < Fq; k0 += 64) {
        __syncthreads();
        if (k0 + 64 < Fq)
            __builtin_prefetch(Wd + (size_t)(k0 + 64 + (tid >> 2)) * Dq + n0, 0, 0);
        // A tile: 128 rows x 64 f16, dense 2D -> TDM (wave 0 issues descriptor)
#ifdef HAVE_TDM
        if (wid == 0)
            tdm_load_tile_2d(A16 + (size_t)m0 * Fq + k0, sA_off, Fq, T);
#else
#pragma unroll
        for (int i = 0; i < 4; ++i) {
            int c   = tid + 256 * i;
            int row = c >> 3;
            int ko  = (c & 7) * 8;
            copy_b128_to_lds(A16 + (size_t)(m0 + row) * Fq + k0 + ko, &sA[row][ko]);
        }
#endif
#pragma unroll
        for (int i = 0; i < 4; ++i) {
            int c  = tid + 256 * i;
            int kk = c >> 4;
            int nc = (c & 15) * 4;
            size_t off = (size_t)(k0 + kk) * Dq + n0 + nc;
            v4f d4 = *(const v4f*)(Wd + off);
#pragma unroll
            for (int j = 0; j < 4; ++j)
                sB[nc + j][kk] = (_Float16)d4[j];
        }
#ifdef HAVE_TDM
        if (wid == 0) wait_tensor_copies();
#else
        wait_async_copies();
#endif
        __syncthreads();

#pragma unroll
        for (int ks = 0; ks < 64; ks += 32) {
            v16h af[2], bf[2];
#pragma unroll
            for (int tm = 0; tm < 2; ++tm) {
                int row = wm + tm * 16 + (lane & 15);
                int kh  = ks + (lane >> 4) * 16;
                v8h lo = *(const v8h*)&sA[row][kh];
                v8h hi = *(const v8h*)&sA[row][kh + 8];
                v16h a;
#pragma unroll
                for (int j = 0; j < 8; ++j) { a[j] = lo[j]; a[j + 8] = hi[j]; }
                af[tm] = a;
            }
#pragma unroll
            for (int tn = 0; tn < 2; ++tn) {
                int nn = wn + tn * 16 + (lane & 15);
                int kh = ks + (lane >> 4) * 16;
                bf[tn] = ld_v16h(&sB[nn][kh]);
            }
#pragma unroll
            for (int tm = 0; tm < 2; ++tm)
#pragma unroll
                for (int tn = 0; tn < 2; ++tn)
                    cc[tm][tn] = __builtin_amdgcn_wmma_f32_16x16x32_f16(
                        false, af[tm], false, bf[tn], (short)0, cc[tm][tn], false, false);
        }
    }

#pragma unroll
    for (int tm = 0; tm < 2; ++tm)
#pragma unroll
        for (int tn = 0; tn < 2; ++tn)
#pragma unroll
            for (int r = 0; r < 8; ++r) {
                int rloc = wm + tm * 16 + (lane >> 4) * 8 + r;
                int tok  = sTok[rloc];
                if (tok >= 0) {
                    int col = n0 + wn + tn * 16 + (lane & 15);
                    float v = cc[tm][tn][r] * sScl[rloc];
                    size_t o = (size_t)tok * Dq + col;
                    if (init) out[o] = v;
                    else      out[o] += v;
                }
            }
}

// ---------------------------------------------------------------------------
extern "C" void kernel_launch(void* const* d_in, const int* in_sizes, int n_in,
                              void* d_out, int out_size, void* d_ws, size_t ws_size,
                              hipStream_t stream) {
    (void)in_sizes; (void)n_in; (void)out_size; (void)ws_size;
    const float* x    = (const float*)d_in[0];
    const float* Wg_s = (const float*)d_in[1];
    const float* Wu_s = (const float*)d_in[2];
    const float* Wd_s = (const float*)d_in[3];
    const float* Wg   = (const float*)d_in[4];
    const float* Wu   = (const float*)d_in[5];
    const float* Wd   = (const float*)d_in[6];
    const float* Wr   = (const float*)d_in[7];
    const float* bias = (const float*)d_in[8];
    float* out = (float*)d_out;

    // workspace: wgt [T,E] f32 | cnt [16] i32 | idx [E,T] i32 | x16 | act (~17MB)
    float*    wgt = (float*)d_ws;
    int*      cnt = (int*)(wgt + (size_t)T * E);
    int*      idx = cnt + 16;
    _Float16* X16 = (_Float16*)(idx + (size_t)E * T);
    _Float16* act = X16 + (size_t)T * Dq;

    cast_x_kernel<<<T * Dq / 4 / 256, 256, 0, stream>>>(x, X16, T * Dq / 4);
    router_kernel<<<T / 8, 256, 0, stream>>>(x, Wr, bias, wgt, out + (size_t)T * Dq);
    build_lists_kernel<<<E, 256, 0, stream>>>(wgt, cnt, idx);

    dim3 blk(256);
    dim3 g1(Fq / 64, T / 128);
    dim3 g2(Dq / 64, T / 128);

    // shared expert (dense, initializes out)
    gemm_gateup_kernel<<<g1, blk, 0, stream>>>(X16, Wg_s, Wu_s, act, nullptr, nullptr);
    gemm_down_kernel<<<g2, blk, 0, stream>>>(act, Wd_s, wgt, -1, nullptr, nullptr, out, 1);

    // routed experts: gathered tokens only (top-2/8 of the dense FLOPs),
    // sequential launches => deterministic accumulation order.
    for (int e = 0; e < E; ++e) {
        gemm_gateup_kernel<<<g1, blk, 0, stream>>>(
            X16, Wg + (size_t)e * Dq * Fq, Wu + (size_t)e * Dq * Fq,
            act, idx + (size_t)e * T, cnt + e);
        gemm_down_kernel<<<g2, blk, 0, stream>>>(
            act, Wd + (size_t)e * Fq * Dq, wgt, e,
            idx + (size_t)e * T, cnt + e, out, 0);
    }
}
